// CM_TPM_88424786690567
// MI455X (gfx1250) — matrix operand order repaired and checked by reading, not compiled, not run on hardware
//
#include <hip/hip_runtime.h>

typedef __attribute__((ext_vector_type(16))) _Float16 v16h;
typedef __attribute__((ext_vector_type(8)))  _Float16 v8h;
typedef __attribute__((ext_vector_type(4)))  _Float16 v4h;
typedef __attribute__((ext_vector_type(8)))  float    v8f;
typedef __attribute__((ext_vector_type(4)))  float    v4f;

#define B_TOT 16384
#define D_DIM 1024
#define K_COMP 512
#define L_DIM 128
#define H_DIM 64

#define BN 64            // batch columns per workgroup
#define LDS_STRIDE 1032  // halves per LDS row: 1024 + 8-half pad (4 dwords) -> conflict-free column reads

// ---------------------------------------------------------------------------
// Kernel 1: phi = relu(z @ W1 + b1) @ W2 + b2  (K x D), emitted as f16,
//           plus phi2[k] = ||phi_k||^2 computed from the f16-rounded values.
// One block per component k, 64 threads (= H).
// ---------------------------------------------------------------------------
__global__ __launch_bounds__(64)
void phinet_kernel(const float* __restrict__ z,
                   const float* __restrict__ W1,
                   const float* __restrict__ b1,
                   const float* __restrict__ W2,
                   const float* __restrict__ b2,
                   _Float16* __restrict__ phi_h,
                   float* __restrict__ phi2) {
  __shared__ float zrow[L_DIM];
  __shared__ float hrow[H_DIM];
  __shared__ float red[H_DIM];
  const int k = blockIdx.x;
  const int t = threadIdx.x;  // 0..63

  zrow[t]      = z[k * L_DIM + t];
  zrow[t + 64] = z[k * L_DIM + t + 64];
  __syncthreads();

  float acc = b1[t];
#pragma unroll 8
  for (int l = 0; l < L_DIM; ++l) acc += zrow[l] * W1[l * H_DIM + t];
  hrow[t] = fmaxf(acc, 0.0f);
  __syncthreads();

  float ss = 0.0f;
  for (int i = 0; i < D_DIM / H_DIM; ++i) {
    const int d = t + i * H_DIM;  // coalesced across t
    float p = b2[d];
#pragma unroll 8
    for (int h = 0; h < H_DIM; ++h) p += hrow[h] * W2[h * D_DIM + d];
    const _Float16 ph = (_Float16)p;
    phi_h[k * D_DIM + d] = ph;
    const float pf = (float)ph;
    ss += pf * pf;
  }
  red[t] = ss;
  __syncthreads();
  if (t == 0) {
    float s = 0.0f;
    for (int i = 0; i < H_DIM; ++i) s += red[i];
    phi2[k] = s;
  }
}

// ---------------------------------------------------------------------------
// Kernel 2: per block of 64 batch columns, compute
//   partial[block] = sum_b log( (1/K) * sum_k exp(2*phi_k.x_b - |phi_k|^2 - |x_b|^2) + eps )
// GEMM core: v_wmma_f32_16x16x32_f16.
//   A = phi tile (global, L2-resident) -- loaded ONCE per dt and reused for
//       two B fragments (two 16-wide column tiles) -> 1 global b128 per WMMA.
//   B = x tile (LDS f16, padded stride -> conflict-free).
// Wave layout: 2 column groups (32 cols) x 4 k groups (8 k-tiles of 16).
// ---------------------------------------------------------------------------
__global__ __launch_bounds__(256)
void mixture_kernel(const float* __restrict__ x,
                    const _Float16* __restrict__ phi_h,
                    const float* __restrict__ phi2,
                    float* __restrict__ partials) {
  __shared__ _Float16 xlds[BN * LDS_STRIDE];
  __shared__ float x2s[BN];
  __shared__ float colsum[4 * BN];

  const int tid  = threadIdx.x;
  const int lane = tid & 31;
  const int wave = tid >> 5;
  const int b0   = blockIdx.x * BN;

  // ---- stage x block into LDS as f16; compute ||x_b||^2 in f32 on the way in
  {
    const int row = tid >> 2;       // 0..63 : batch row within block
    const int q   = tid & 3;        // quarter of the row
    const int d0  = q * 256;
    const float* __restrict__ src = x + (size_t)(b0 + row) * D_DIM + d0;
    _Float16* __restrict__ dst = &xlds[row * LDS_STRIDE + d0];
    float ss = 0.0f;
#pragma unroll 4
    for (int i = 0; i < 64; ++i) {
      const v4f v = *(const v4f*)(src + i * 4);
      ss += v.x * v.x + v.y * v.y + v.z * v.z + v.w * v.w;
      v4h hv;
      hv.x = (_Float16)v.x; hv.y = (_Float16)v.y;
      hv.z = (_Float16)v.z; hv.w = (_Float16)v.w;
      *(v4h*)(dst + i * 4) = hv;
    }
    ss += __shfl_xor(ss, 1, 32);
    ss += __shfl_xor(ss, 2, 32);
    if (q == 0) x2s[row] = ss;
  }
  __syncthreads();

  // ---- wave assignment: 2 waves across columns x 4 waves across k-tiles
  const int wb   = wave & 1;        // column group: cols [wb*32, wb*32+32)
  const int wk   = wave >> 1;       // k group: k-tiles [wk*8, wk*8+8)
  const bool hi  = lane >= 16;
  const int col0 = (wb << 5) + (lane & 15);   // first column tile (C/D: N = lane%16)
  const int col1 = col0 + 16;                 // second column tile
  const int mrow = lane & 15;                 // A-matrix row M = lane%16
  const int aoff0 = hi ? 16 : 0;              // A lane<16: K0..15;        lane>=16: K16..31
  const int boff  = hi ? 16 : 0;              // B lane<16: K0..15;        lane>=16: K16..31

  const float x2v0 = x2s[col0];
  const float x2v1 = x2s[col1];
  const _Float16* __restrict__ xrow0 = &xlds[col0 * LDS_STRIDE];
  const _Float16* __restrict__ xrow1 = &xlds[col1 * LDS_STRIDE];

  float s0 = 0.0f, s1 = 0.0f;
  for (int kt = wk * 8; kt < wk * 8 + 8; ++kt) {
    const int k0 = kt << 4;
    const _Float16* __restrict__ arow = phi_h + (size_t)(k0 + mrow) * D_DIM;
    v8f acc0 = {};
    v8f acc1 = {};
#pragma unroll 4
    for (int dt = 0; dt < 32; ++dt) {
      const int d0 = dt << 5;
      union { v16h v; v8h h[2]; } a, b0v, b1v;
      a.h[0]   = *(const v8h*)(arow + d0 + aoff0);
      a.h[1]   = *(const v8h*)(arow + d0 + aoff0 + 8);
      b0v.h[0] = *(const v8h*)(xrow0 + d0 + boff);
      b0v.h[1] = *(const v8h*)(xrow0 + d0 + boff + 8);
      b1v.h[0] = *(const v8h*)(xrow1 + d0 + boff);
      b1v.h[1] = *(const v8h*)(xrow1 + d0 + boff + 8);
      acc0 = __builtin_amdgcn_wmma_f32_16x16x32_f16(
          false, a.v, false, b0v.v, (short)0, acc0, false, false);
      acc1 = __builtin_amdgcn_wmma_f32_16x16x32_f16(
          false, a.v, false, b1v.v, (short)0, acc1, false, false);
    }
    // epilogue: accN[r] = phi[k0 + r + (hi?8:0)] . x[colN]
    const int kbase = k0 + (hi ? 8 : 0);
#pragma unroll
    for (int r = 0; r < 8; ++r) {
      const float p2 = phi2[kbase + r];
      const float e0 = 2.0f * acc0[r] - p2 - x2v0;  // = -||phi-x||^2
      const float e1 = 2.0f * acc1[r] - p2 - x2v1;
      s0 += __builtin_amdgcn_exp2f(e0 * 1.442695040888963f);
      s1 += __builtin_amdgcn_exp2f(e1 * 1.442695040888963f);
    }
  }

  // lanes c and c+16 hold the same column (different k rows of the tile)
  s0 += __shfl_xor(s0, 16, 32);
  s1 += __shfl_xor(s1, 16, 32);
  if (!hi) {
    colsum[wk * BN + col0] = s0;
    colsum[wk * BN + col1] = s1;
  }
  __syncthreads();

  if (tid < BN) {
    const float tot = ((colsum[tid] + colsum[BN + tid]) +
                       (colsum[2 * BN + tid] + colsum[3 * BN + tid]));
    colsum[tid] =
        __builtin_amdgcn_logf(tot * (1.0f / (float)K_COMP) + 1e-9f) * 0.6931471805599453f;
  }
  __syncthreads();
  if (tid == 0) {
    float t = 0.0f;
    for (int i = 0; i < BN; ++i) t += colsum[i];  // fixed order: deterministic
    partials[blockIdx.x] = t;
  }
}

// ---------------------------------------------------------------------------
// Kernel 3: deterministic final reduction -> scalar mean over B
// ---------------------------------------------------------------------------
__global__ void finalize_kernel(const float* __restrict__ partials,
                                float* __restrict__ out, int nparts) {
  if (threadIdx.x == 0 && blockIdx.x == 0) {
    float s = 0.0f;
    for (int i = 0; i < nparts; ++i) s += partials[i];
    out[0] = s * (1.0f / (float)B_TOT);
  }
}

extern "C" void kernel_launch(void* const* d_in, const int* in_sizes, int n_in,
                              void* d_out, int out_size, void* d_ws, size_t ws_size,
                              hipStream_t stream) {
  (void)in_sizes; (void)n_in; (void)out_size; (void)ws_size;
  const float* x  = (const float*)d_in[0];
  const float* z  = (const float*)d_in[1];
  const float* W1 = (const float*)d_in[2];
  const float* b1 = (const float*)d_in[3];
  const float* W2 = (const float*)d_in[4];
  const float* b2 = (const float*)d_in[5];

  char* ws = (char*)d_ws;
  _Float16* phi_h  = (_Float16*)ws;                      // 512*1024*2 = 1 MiB
  float*    phi2   = (float*)(ws + (1 << 20));           // 2 KiB
  float*    parts  = (float*)(ws + (1 << 20) + 4096);    // 1 KiB

  phinet_kernel<<<K_COMP, 64, 0, stream>>>(z, W1, b1, W2, b2, phi_h, phi2);
  mixture_kernel<<<B_TOT / BN, 256, 0, stream>>>(x, phi_h, phi2, parts);
  finalize_kernel<<<1, 32, 0, stream>>>(parts, (float*)d_out, B_TOT / BN);
}
